// MoE_25151328485988
// MI455X (gfx1250) — matrix operand order repaired and checked by reading, not compiled, not run on hardware
//
#include <hip/hip_runtime.h>

#define Dd 1024
#define Hh 2048
#define Nn 8
#define Kk 2
#define Mm 8192   // B*S = 4*2048

typedef unsigned short u16;
typedef __attribute__((ext_vector_type(4)))  unsigned int u32x4;
typedef __attribute__((ext_vector_type(16))) __bf16       v16bf;
typedef __attribute__((ext_vector_type(8)))  float        v8f;

union Frag { u32x4 q[2]; v16bf v; };

__device__ __forceinline__ u16 f2bf(float f) {
  unsigned u = __float_as_uint(f);
  u += 0x7fffu + ((u >> 16) & 1u);        // round-to-nearest-even
  return (u16)(u >> 16);
}

// LDS aperture: generic pointer low 32 bits are the LDS byte address (ISA 10.2)
__device__ __forceinline__ unsigned lds_off_of(const void* p) {
  return (unsigned)(size_t)p;
}

// CDNA5 async global->LDS copy, 16B per lane, tracked with ASYNCcnt
__device__ __forceinline__ void async_ld16(const u16* gptr, unsigned lds_off) {
  asm volatile("global_load_async_to_lds_b128 %0, %1, off"
               :: "v"(lds_off), "v"((unsigned long long)(size_t)gptr)
               : "memory");
}
__device__ __forceinline__ void wait_async0() {
  asm volatile("s_wait_asynccnt 0x0" ::: "memory");
}

// ---------------- small utility kernels ----------------

__global__ void zero_counts_kernel(int* counts) {
  if (threadIdx.x < Nn) counts[threadIdx.x] = 0;
}

__global__ __launch_bounds__(256) void conv_x_kernel(const float* __restrict__ x,
                                                     u16* __restrict__ xb) {
  int i = blockIdx.x * 256 + threadIdx.x;
  xb[i] = f2bf(x[i]);
}

// Wl1: [N][D][2H] fp32  ->  w1t: [N][2H][D] bf16
__global__ __launch_bounds__(256) void conv_w1t_kernel(const float* __restrict__ Wl1,
                                                       u16* __restrict__ w1t) {
  size_t i = (size_t)blockIdx.x * 256 + threadIdx.x;
  size_t per = (size_t)Dd * (2 * Hh);
  int n = (int)(i / per);
  size_t rem = i - (size_t)n * per;
  int h = (int)(rem / Dd);
  int d = (int)(rem - (size_t)h * Dd);
  w1t[i] = f2bf(Wl1[((size_t)n * Dd + d) * (2 * Hh) + h]);
}

// Wl2: [N][H][D] fp32  ->  w2t: [N][D][H] bf16
__global__ __launch_bounds__(256) void conv_w2t_kernel(const float* __restrict__ Wl2,
                                                       u16* __restrict__ w2t) {
  size_t i = (size_t)blockIdx.x * 256 + threadIdx.x;
  size_t per = (size_t)Dd * Hh;
  int n = (int)(i / per);
  size_t rem = i - (size_t)n * per;
  int d = (int)(rem / Hh);
  int h = (int)(rem - (size_t)d * Hh);
  w2t[i] = f2bf(Wl2[((size_t)n * Hh + h) * Dd + d]);
}

// ---------------- router ----------------

__global__ __launch_bounds__(256) void router_kernel(
    const float* __restrict__ x, const float* __restrict__ Wr,
    const float* __restrict__ br, const float* __restrict__ biasN,
    int* __restrict__ idx_MK, float* __restrict__ val_MK,
    int* __restrict__ pos_MK, int* __restrict__ counts)
{
  int lane = threadIdx.x & 31;
  int wave = threadIdx.x >> 5;
  int m = blockIdx.x * 8 + wave;
  if (m >= Mm) return;

  float s[Nn];
#pragma unroll
  for (int n = 0; n < Nn; ++n) s[n] = 0.f;

  const float* xr = x + (size_t)m * Dd;
  for (int j = lane; j < Dd; j += 32) {
    float xv = xr[j];
#pragma unroll
    for (int n = 0; n < Nn; ++n) s[n] += xv * Wr[n * Dd + j];
  }
#pragma unroll
  for (int n = 0; n < Nn; ++n) {
#pragma unroll
    for (int off = 16; off > 0; off >>= 1) s[n] += __shfl_xor(s[n], off, 32);
  }

  if (lane == 0) {
    float sc[Nn], w[Nn];
    float mx = -1e30f;
#pragma unroll
    for (int n = 0; n < Nn; ++n) { sc[n] = s[n] + br[n]; mx = fmaxf(mx, sc[n]); }
    float sum = 0.f;
#pragma unroll
    for (int n = 0; n < Nn; ++n) { w[n] = __expf(sc[n] - mx); sum += w[n]; }
    float inv = 1.f / sum;
#pragma unroll
    for (int n = 0; n < Nn; ++n) w[n] *= inv;

    int i0 = 0, i1 = 0; float b0 = -1e30f, b1 = -1e30f;
#pragma unroll
    for (int n = 0; n < Nn; ++n) {
      float k = sc[n] + biasN[n];
      if (k > b0)      { b1 = b0; i1 = i0; b0 = k; i0 = n; }
      else if (k > b1) { b1 = k; i1 = n; }
    }
    float v0 = w[i0], v1 = w[i1];
    float vs = 1.f / (v0 + v1);
    v0 *= vs; v1 *= vs;

    int p0 = atomicAdd(&counts[i0], 1);
    int p1 = atomicAdd(&counts[i1], 1);
    idx_MK[m * 2]     = i0;  idx_MK[m * 2 + 1] = i1;
    val_MK[m * 2]     = v0;  val_MK[m * 2 + 1] = v1;
    pos_MK[m * 2]     = p0;  pos_MK[m * 2 + 1] = p1;
  }
}

__global__ void offsets_kernel(const int* __restrict__ counts, int* __restrict__ offs) {
  if (threadIdx.x == 0 && blockIdx.x == 0) {
    int a = 0;
    for (int n = 0; n < Nn; ++n) { offs[n] = a; a += counts[n]; }
  }
}

__global__ __launch_bounds__(256) void scatter_kernel(
    const int* __restrict__ idx_MK, const int* __restrict__ pos_MK,
    const int* __restrict__ offs, int* __restrict__ slot_MK,
    int* __restrict__ rowlist)
{
  int i = blockIdx.x * 256 + threadIdx.x;
  if (i >= Mm * Kk) return;
  int e = idx_MK[i];
  int slot = offs[e] + pos_MK[i];
  slot_MK[i] = slot;
  rowlist[slot] = i >> 1;           // token id
}

// ---------------- expert GEMM1: gather(x) @ W1 (both halves), SwiGLU -> g ----------------
// block tile: 128 rows x 64 cols (per gate/up half), K step 32, 8 waves of 32,
// double-buffered LDS fed by async global->LDS copies.

__global__ __launch_bounds__(256) void gemm1_kernel(
    const u16* __restrict__ xb, const u16* __restrict__ w1t,
    const float* __restrict__ bl1,
    const int* __restrict__ rowlist, const int* __restrict__ offs,
    const int* __restrict__ counts, u16* __restrict__ g)
{
  const int e = blockIdx.z;
  const int Me = counts[e];
  const int row0 = blockIdx.y * 128;
  if (row0 >= Me) return;
  const int col0 = blockIdx.x * 64;                 // within [0,H)
  const u16* w1e = w1t + (size_t)e * (2 * Hh) * Dd; // [2H][D]
  const int base = offs[e];

  __shared__ u16 As [2][128][40];
  __shared__ u16 Bat[2][64][40];
  __shared__ u16 Bbt[2][64][40];

  const int tid  = threadIdx.x;
  const int lane = tid & 31;
  const int wave = tid >> 5;
  const int wr = wave >> 1, wc = wave & 1;
  const int lrow = lane & 15;
  const int hi = lane >> 4;                         // 0 or 1

  // A-gather: each thread stages 2 x 8-bf16 chunks of one row
  const int ar  = tid >> 1;                         // 0..127
  const int akc = (tid & 1) * 16;                   // 0 or 16
  int arow = row0 + ar;
  int arc  = (arow < Me) ? arow : (Me - 1);
  const u16* axp = xb + (size_t)rowlist[base + arc] * Dd;

  // B loads: each thread one 8-bf16 chunk per half
  const int bh  = tid >> 2;                         // 0..63  (column within tile)
  const int bkc = (tid & 3) * 8;                    // 0,8,16,24
  const u16* bap = w1e + (size_t)(col0 + bh) * Dd;
  const u16* bbp = w1e + (size_t)(Hh + col0 + bh) * Dd;

  unsigned lA0[2], lA1[2], lBa[2], lBb[2];
#pragma unroll
  for (int b = 0; b < 2; ++b) {
    lA0[b] = lds_off_of(&As[b][ar][akc]);
    lA1[b] = lds_off_of(&As[b][ar][akc + 8]);
    lBa[b] = lds_off_of(&Bat[b][bh][bkc]);
    lBb[b] = lds_off_of(&Bbt[b][bh][bkc]);
  }

  v8f accA[2][2] = {};
  v8f accB[2][2] = {};

  // prologue: stage tile 0 into buffer 0
  async_ld16(axp + 0 + akc,     lA0[0]);
  async_ld16(axp + 0 + akc + 8, lA1[0]);
  async_ld16(bap + 0 + bkc,     lBa[0]);
  async_ld16(bbp + 0 + bkc,     lBb[0]);

  for (int k0 = 0; k0 < Dd; k0 += 32) {
    const int b = (k0 >> 5) & 1;
    wait_async0();        // my tile-k0 copies landed in LDS
    __syncthreads();      // everyone's landed; everyone done reading buffer b^1

    if (k0 + 32 < Dd) {   // stream next tile into the other buffer
      async_ld16(axp + k0 + 32 + akc,     lA0[b ^ 1]);
      async_ld16(axp + k0 + 32 + akc + 8, lA1[b ^ 1]);
      async_ld16(bap + k0 + 32 + bkc,     lBa[b ^ 1]);
      async_ld16(bbp + k0 + 32 + bkc,     lBb[b ^ 1]);
    }

    Frag af[2], bfa[2], bfb[2];
#pragma unroll
    for (int mi = 0; mi < 2; ++mi) {
      const u16* p = &As[b][wr * 32 + mi * 16 + lrow][hi * 16];
      af[mi].q[0] = *(const u32x4*)(p);
      af[mi].q[1] = *(const u32x4*)(p + 8);
    }
#pragma unroll
    for (int ni = 0; ni < 2; ++ni) {
      const u16* pa = &Bat[b][wc * 32 + ni * 16 + lrow][hi * 16];
      bfa[ni].q[0] = *(const u32x4*)(pa);
      bfa[ni].q[1] = *(const u32x4*)(pa + 8);
      const u16* pb = &Bbt[b][wc * 32 + ni * 16 + lrow][hi * 16];
      bfb[ni].q[0] = *(const u32x4*)(pb);
      bfb[ni].q[1] = *(const u32x4*)(pb + 8);
    }
#pragma unroll
    for (int mi = 0; mi < 2; ++mi)
#pragma unroll
      for (int ni = 0; ni < 2; ++ni) {
        accA[mi][ni] = __builtin_amdgcn_wmma_f32_16x16x32_bf16(
            false, af[mi].v, false, bfa[ni].v, (short)0, accA[mi][ni], false, false);
        accB[mi][ni] = __builtin_amdgcn_wmma_f32_16x16x32_bf16(
            false, af[mi].v, false, bfb[ni].v, (short)0, accB[mi][ni], false, false);
      }
  }

#pragma unroll
  for (int mi = 0; mi < 2; ++mi)
#pragma unroll
    for (int ni = 0; ni < 2; ++ni) {
      int colg = col0 + wc * 32 + ni * 16 + lrow;   // within [0,H)
      float ba = bl1[colg];
      float bb = bl1[Hh + colg];
#pragma unroll
      for (int v = 0; v < 8; ++v) {
        int rg = row0 + wr * 32 + mi * 16 + v + hi * 8;
        if (rg < Me) {
          float a = accA[mi][ni][v] + ba;
          float b2 = accB[mi][ni][v] + bb;
          float sig = 1.0f / (1.0f + __expf(-a));
          g[(size_t)(base + rg) * Hh + colg] = f2bf(a * sig * b2);
        }
      }
    }
}

// ---------------- expert GEMM2: g @ W2 + bl2 -> yslot (f32) ----------------

__global__ __launch_bounds__(256) void gemm2_kernel(
    const u16* __restrict__ g, const u16* __restrict__ w2t,
    const float* __restrict__ bl2,
    const int* __restrict__ offs, const int* __restrict__ counts,
    float* __restrict__ yslot)
{
  const int e = blockIdx.z;
  const int Me = counts[e];
  const int row0 = blockIdx.y * 128;
  if (row0 >= Me) return;
  const int col0 = blockIdx.x * 64;                 // within [0,D)
  const u16* w2e = w2t + (size_t)e * Dd * Hh;       // [D][H]
  const int base = offs[e];

  __shared__ u16 As[2][128][40];
  __shared__ u16 Bt[2][64][40];

  const int tid  = threadIdx.x;
  const int lane = tid & 31;
  const int wave = tid >> 5;
  const int wr = wave >> 1, wc = wave & 1;
  const int lrow = lane & 15;
  const int hi = lane >> 4;

  const int ar  = tid >> 1;
  const int akc = (tid & 1) * 16;
  int arow = row0 + ar;
  int arc  = (arow < Me) ? arow : (Me - 1);
  const u16* axp = g + (size_t)(base + arc) * Hh;

  const int bh  = tid >> 2;
  const int bkc = (tid & 3) * 8;
  const u16* bp = w2e + (size_t)(col0 + bh) * Hh;

  unsigned lA0[2], lA1[2], lB[2];
#pragma unroll
  for (int b = 0; b < 2; ++b) {
    lA0[b] = lds_off_of(&As[b][ar][akc]);
    lA1[b] = lds_off_of(&As[b][ar][akc + 8]);
    lB[b]  = lds_off_of(&Bt[b][bh][bkc]);
  }

  v8f acc[2][2] = {};

  async_ld16(axp + 0 + akc,     lA0[0]);
  async_ld16(axp + 0 + akc + 8, lA1[0]);
  async_ld16(bp + 0 + bkc,      lB[0]);

  for (int k0 = 0; k0 < Hh; k0 += 32) {
    const int b = (k0 >> 5) & 1;
    wait_async0();
    __syncthreads();

    if (k0 + 32 < Hh) {
      async_ld16(axp + k0 + 32 + akc,     lA0[b ^ 1]);
      async_ld16(axp + k0 + 32 + akc + 8, lA1[b ^ 1]);
      async_ld16(bp + k0 + 32 + bkc,      lB[b ^ 1]);
    }

    Frag af[2], bf[2];
#pragma unroll
    for (int mi = 0; mi < 2; ++mi) {
      const u16* p = &As[b][wr * 32 + mi * 16 + lrow][hi * 16];
      af[mi].q[0] = *(const u32x4*)(p);
      af[mi].q[1] = *(const u32x4*)(p + 8);
    }
#pragma unroll
    for (int ni = 0; ni < 2; ++ni) {
      const u16* pb = &Bt[b][wc * 32 + ni * 16 + lrow][hi * 16];
      bf[ni].q[0] = *(const u32x4*)(pb);
      bf[ni].q[1] = *(const u32x4*)(pb + 8);
    }
#pragma unroll
    for (int mi = 0; mi < 2; ++mi)
#pragma unroll
      for (int ni = 0; ni < 2; ++ni)
        acc[mi][ni] = __builtin_amdgcn_wmma_f32_16x16x32_bf16(
            false, af[mi].v, false, bf[ni].v, (short)0, acc[mi][ni], false, false);
  }

#pragma unroll
  for (int mi = 0; mi < 2; ++mi)
#pragma unroll
    for (int ni = 0; ni < 2; ++ni) {
      int colg = col0 + wc * 32 + ni * 16 + lrow;   // within [0,D)
      float bc = bl2[colg];
#pragma unroll
      for (int v = 0; v < 8; ++v) {
        int rg = row0 + wr * 32 + mi * 16 + v + hi * 8;
        if (rg < Me)
          yslot[(size_t)(base + rg) * Dd + colg] = acc[mi][ni][v] + bc;
      }
    }
}

// ---------------- combine ----------------

__global__ __launch_bounds__(256) void combine_kernel(
    const float* __restrict__ yslot, const int* __restrict__ slot_MK,
    const float* __restrict__ val_MK, float* __restrict__ out)
{
  int i = blockIdx.x * 256 + threadIdx.x;
  if (i >= Mm * Dd) return;
  int m = i >> 10;            // / D (=1024)
  int d = i & (Dd - 1);
  int s0 = slot_MK[m * 2], s1 = slot_MK[m * 2 + 1];
  float v0 = val_MK[m * 2], v1 = val_MK[m * 2 + 1];
  out[i] = v0 * yslot[(size_t)s0 * Dd + d] + v1 * yslot[(size_t)s1 * Dd + d];
}

// ---------------- launcher ----------------

extern "C" void kernel_launch(void* const* d_in, const int* in_sizes, int n_in,
                              void* d_out, int out_size, void* d_ws, size_t ws_size,
                              hipStream_t stream) {
  (void)in_sizes; (void)n_in; (void)out_size; (void)ws_size;
  const float* x     = (const float*)d_in[0];
  const float* Wr    = (const float*)d_in[1];
  const float* br    = (const float*)d_in[2];
  const float* Wl1   = (const float*)d_in[3];
  const float* bl1   = (const float*)d_in[4];
  const float* Wl2   = (const float*)d_in[5];
  const float* bl2   = (const float*)d_in[6];
  const float* biasN = (const float*)d_in[7];
  float* out = (float*)d_out;

  char* ws = (char*)d_ws;
  size_t off = 0;
  auto alloc = [&](size_t bytes) -> void* {
    void* p = ws + off;
    off = (off + bytes + 255) & ~(size_t)255;
    return p;
  };
  u16*   xb      = (u16*)  alloc((size_t)Mm * Dd * 2);
  u16*   w1t     = (u16*)  alloc((size_t)Nn * 2 * Hh * Dd * 2);
  u16*   w2t     = (u16*)  alloc((size_t)Nn * Dd * Hh * 2);
  u16*   gbuf    = (u16*)  alloc((size_t)Mm * Kk * Hh * 2);
  float* yslot   = (float*)alloc((size_t)Mm * Kk * Dd * 4);
  int*   idx_MK  = (int*)  alloc((size_t)Mm * Kk * 4);
  float* val_MK  = (float*)alloc((size_t)Mm * Kk * 4);
  int*   pos_MK  = (int*)  alloc((size_t)Mm * Kk * 4);
  int*   slot_MK = (int*)  alloc((size_t)Mm * Kk * 4);
  int*   rowlist = (int*)  alloc((size_t)Mm * Kk * 4);
  int*   counts  = (int*)  alloc(Nn * 4);
  int*   offs    = (int*)  alloc(Nn * 4);

  zero_counts_kernel<<<1, 32, 0, stream>>>(counts);
  conv_x_kernel  <<<(Mm * Dd) / 256, 256, 0, stream>>>(x, xb);
  conv_w1t_kernel<<<(Nn * 2 * Hh * Dd) / 256, 256, 0, stream>>>(Wl1, w1t);
  conv_w2t_kernel<<<(Nn * Dd * Hh) / 256, 256, 0, stream>>>(Wl2, w2t);
  router_kernel  <<<Mm / 8, 256, 0, stream>>>(x, Wr, br, biasN, idx_MK, val_MK, pos_MK, counts);
  offsets_kernel <<<1, 1, 0, stream>>>(counts, offs);
  scatter_kernel <<<(Mm * Kk) / 256, 256, 0, stream>>>(idx_MK, pos_MK, offs, slot_MK, rowlist);
  gemm1_kernel<<<dim3(Hh / 64, Mm / 128, Nn), 256, 0, stream>>>(xb, w1t, bl1, rowlist, offs, counts, gbuf);
  gemm2_kernel<<<dim3(Dd / 64, Mm / 128, Nn), 256, 0, stream>>>(gbuf, w2t, bl2, offs, counts, yslot);
  combine_kernel<<<(Mm * Dd) / 256, 256, 0, stream>>>(yslot, slot_MK, val_MK, out);
}
